// GridSampleCrossBEVAttention_naviscore_67740224193153
// MI455X (gfx1250) — compile-verified, matrix operand order checked by reading, not run on hardware
//
#include <hip/hip_runtime.h>
#include <math.h>

// ---------------------------------------------------------------------------
// GridSampleCrossBEVAttention (eval branch) for MI455X / gfx1250.
//
// Algebraic collapse: the grid point, the score scalar, and hence the whole
// attention output are constant across the NQ=1280 query axis:
//   v_b[c]   = s_b * bilinear( relu(conv3x3(bev)[b, c, @navi_pt]) )   (256)
//   w_b[d]   = sum_c out_w[d,c] * v_b[c] + out_b[d]                   (256)
//   out[b,q] = queries[b,q] + w_b                                     (stream)
// Stage 3 is the only HBM-bound part (~21 MB => ~1 us at 23.3 TB/s).
// Stage 2 uses v_wmma_f32_16x16x32_f16 with branch-free b128 operand loads.
// ---------------------------------------------------------------------------

#define NQ   1280
#define DIM  256
#define CIN  64

typedef __attribute__((ext_vector_type(16))) _Float16 v16h;
typedef __attribute__((ext_vector_type(8)))  float    v8f;

// ---------------------------------------------------------------------------
// Stage 1: one block per batch. Computes s_b (sine-embed score dot aws_w),
// loads the 4x4 x 64ch input patch around the bilinear taps into LDS, runs the
// 3x3 conv for its 256 output channels at the 4 tap pixels, applies bias+ReLU,
// bilinear-combines with zero padding, scales by s_b -> vin[b*256 + oc].
// ---------------------------------------------------------------------------
__global__ __launch_bounds__(256) void gscba_stage1_sample(
    const float* __restrict__ navi,    // (B,2)
    const float* __restrict__ bev,     // (B,64,H,W)
    const int*   __restrict__ shp,     // [H,W]
    const float* __restrict__ pscore,  // (B,2)
    const float* __restrict__ aws_w,   // (1,256)
    const float* __restrict__ aws_b,   // (1,)
    const float* __restrict__ conv_w,  // (256,64,3,3)
    const float* __restrict__ conv_b,  // (256,)
    float*       __restrict__ vin)     // (B,256) workspace
{
  const int b = blockIdx.x;
  const int t = threadIdx.x;            // output channel 0..255
  const int H = shp[0], W = shp[1];

  __shared__ float red[256];
  __shared__ float patch[CIN * 16];     // 64ch x 4x4 patch

  // ---- s_b = sineembed(point_score[b]) . aws_w + aws_b  (hidden_dim=256)
  {
    // emb[0:128] from point_score[...,1] (py), emb[128:256] from [...,0] (px)
    const float val = (t < 128) ? pscore[b * 2 + 1] : pscore[b * 2 + 0];
    const int   dd  = (t < 128) ? t : (t - 128);
    const int   j   = dd >> 1;                       // dim_t index pair
    const float dim_t = powf(10000.0f, (float)j / 64.0f);
    const float p = val * 6.283185307179586f / dim_t;
    const float e = (dd & 1) ? cosf(p) : sinf(p);
    red[t] = e * aws_w[t];
  }
  __syncthreads();
  for (int off = 128; off > 0; off >>= 1) {
    if (t < off) red[t] += red[t + off];
    __syncthreads();
  }
  const float s_b = red[0] + aws_b[0];

  // ---- bilinear sample coords: grid = (navi_y/32, navi_x/32), align_corners=F
  const float nx = navi[b * 2 + 0];
  const float ny = navi[b * 2 + 1];
  const float gx = ((ny * (1.0f / 32.0f)) + 1.0f) * 0.5f * (float)W - 0.5f;
  const float gy = ((nx * (1.0f / 32.0f)) + 1.0f) * 0.5f * (float)H - 0.5f;
  const float x0f = floorf(gx), y0f = floorf(gy);
  const int   x0 = (int)x0f,  y0 = (int)y0f;
  const float wx1 = gx - x0f, wy1 = gy - y0f;

  float tw[4];
  tw[0] = (1.0f - wx1) * (1.0f - wy1);   // (x0  , y0  )
  tw[1] =         wx1  * (1.0f - wy1);   // (x0+1, y0  )
  tw[2] = (1.0f - wx1) *         wy1;    // (x0  , y0+1)
  tw[3] =         wx1  *         wy1;    // (x0+1, y0+1)
  const int txs[4] = {x0, x0 + 1, x0, x0 + 1};
  const int tys[4] = {y0, y0, y0 + 1, y0 + 1};
  #pragma unroll
  for (int tp = 0; tp < 4; ++tp) {
    const bool valid = (txs[tp] >= 0) && (txs[tp] < W) &&
                       (tys[tp] >= 0) && (tys[tp] < H);
    if (!valid) tw[tp] = 0.0f;           // padding_mode='zeros'
  }

  // ---- cooperative patch load (rows y0-1..y0+2, cols x0-1..x0+2, zero-pad)
  #pragma unroll
  for (int k = 0; k < 4; ++k) {
    const int lin = t + k * 256;         // 0..1023
    const int ic  = lin >> 4;
    const int pos = lin & 15;
    const int r = pos >> 2, c = pos & 3;
    const int yy = y0 - 1 + r, xx = x0 - 1 + c;
    float v = 0.0f;
    if (yy >= 0 && yy < H && xx >= 0 && xx < W)
      v = bev[((size_t)(b * CIN + ic) * H + yy) * W + xx];
    patch[lin] = v;
  }
  __syncthreads();

  // ---- 3x3 conv at the 4 tap pixels for out-channel t
  float a0 = 0.f, a1 = 0.f, a2 = 0.f, a3 = 0.f;
  const float* wrow = conv_w + (size_t)t * CIN * 9;
  for (int ic = 0; ic < CIN; ++ic) {
    const float* pch = patch + ic * 16;
    #pragma unroll
    for (int kh = 0; kh < 3; ++kh) {
      #pragma unroll
      for (int kw = 0; kw < 3; ++kw) {
        const float wgt = wrow[ic * 9 + kh * 3 + kw];
        a0 = fmaf(wgt, pch[(kh    ) * 4 + (kw    )], a0);
        a1 = fmaf(wgt, pch[(kh    ) * 4 + (kw + 1)], a1);
        a2 = fmaf(wgt, pch[(kh + 1) * 4 + (kw    )], a2);
        a3 = fmaf(wgt, pch[(kh + 1) * 4 + (kw + 1)], a3);
      }
    }
  }
  const float cb = conv_b[t];
  const float r0 = fmaxf(a0 + cb, 0.0f);
  const float r1 = fmaxf(a1 + cb, 0.0f);
  const float r2 = fmaxf(a2 + cb, 0.0f);
  const float r3 = fmaxf(a3 + cb, 0.0f);
  const float sampled = tw[0] * r0 + tw[1] * r1 + tw[2] * r2 + tw[3] * r3;

  vin[b * DIM + t] = sampled * s_b;      // (aw==1 since softmax over 1 elem)
}

// Pack 8 consecutive f32 (two float4 runs) into fragment halves as f16.
__device__ __forceinline__ void gscba_pack8(v16h& v, int base,
                                            float4 lo, float4 hi) {
  v[base + 0] = (_Float16)lo.x; v[base + 1] = (_Float16)lo.y;
  v[base + 2] = (_Float16)lo.z; v[base + 3] = (_Float16)lo.w;
  v[base + 4] = (_Float16)hi.x; v[base + 5] = (_Float16)hi.y;
  v[base + 6] = (_Float16)hi.z; v[base + 7] = (_Float16)hi.w;
}

// ---------------------------------------------------------------------------
// Stage 2: w[b][d] = sum_c out_w[d][c] * vin[b][c] + out_b[d]
// One wave per 16-row d-tile; v_wmma_f32_16x16x32_f16 over the 256-deep c dim
// (8 K-chunks). Batches occupy the N dimension.
//
// Per the CDNA5 16-bit A 16x32 layout, each lane's fragment is two contiguous
// 8-element K runs: e in [0,8) <-> K = khalf*8 + e, e in [8,16) <-> K = 16 +
// khalf*8 + (e-8). So operands load as four b128 vectors per chunk — no
// scatter. B rows >= nb are clamped (not branched): WMMA column n of C depends
// only on column n of B, and columns >= nb are never stored, so the duplicate
// data is harmless and EXEC stays all-1s with zero divergent branches.
// ---------------------------------------------------------------------------
__global__ __launch_bounds__(32) void gscba_stage2_matvec_wmma(
    const float* __restrict__ out_w,   // (256,256) row-major [d][c]
    const float* __restrict__ out_b,   // (256,)
    const float* __restrict__ vin,     // (B,256)
    float*       __restrict__ wvec,    // (B,256) workspace
    int nb)
{
  const int dtile = blockIdx.x;        // 0..15 -> rows d = dtile*16 + 0..15
  const int lane  = threadIdx.x;       // 0..31 (wave32)
  const int row   = lane & 15;         // A-row (M) / B-col (N)
  const int khalf = lane >> 4;         // K half-select per ISA layout

  const float* arow = out_w + (size_t)(dtile * 16 + row) * DIM;
  const int    brow = (row < nb) ? row : (nb - 1);   // clamp; cols>=nb unused
  const float* bro  = vin + (size_t)brow * DIM;

  v8f c = {};
  #pragma unroll
  for (int kc = 0; kc < DIM / 32; ++kc) {
    const int k0 = kc * 32 + khalf * 8;        // run for elements 0..7
    const int k1 = kc * 32 + 16 + khalf * 8;   // run for elements 8..15

    const float4 aL0 = *(const float4*)(arow + k0);
    const float4 aL1 = *(const float4*)(arow + k0 + 4);
    const float4 aH0 = *(const float4*)(arow + k1);
    const float4 aH1 = *(const float4*)(arow + k1 + 4);
    const float4 bL0 = *(const float4*)(bro + k0);
    const float4 bL1 = *(const float4*)(bro + k0 + 4);
    const float4 bH0 = *(const float4*)(bro + k1);
    const float4 bH1 = *(const float4*)(bro + k1 + 4);

    v16h a, bm;
    gscba_pack8(a,  0, aL0, aL1);
    gscba_pack8(a,  8, aH0, aH1);
    gscba_pack8(bm, 0, bL0, bL1);
    gscba_pack8(bm, 8, bH0, bH1);

    // D = A x B + C  (8 args: neg_a, A, neg_b, B, c_mod, C, reuse_a, reuse_b)
    c = __builtin_amdgcn_wmma_f32_16x16x32_f16(false, a, false, bm,
                                               (short)0, c, false, false);
  }

  // f32 C/D layout: VGPR r -> lanes 0-15: (M=r, N=lane); 16-31: (M=r+8, N=lane-16)
  const int col   = lane & 15;         // batch index N
  const int mbase = (lane >> 4) * 8;
  if (col < nb) {
    #pragma unroll
    for (int r = 0; r < 8; ++r) {
      const int d = dtile * 16 + mbase + r;
      wvec[col * DIM + d] = c[r] + out_b[d];
    }
  }
}

// ---------------------------------------------------------------------------
// Stage 3: out[b,q,:] = queries[b,q,:] + wvec[b,:]  (HBM-bound streaming add)
// float4 vectorized: 128-bit loads/stores; wvec stays L2/L0 resident.
// ---------------------------------------------------------------------------
__global__ __launch_bounds__(256) void gscba_stage3_add(
    const float* __restrict__ q, const float* __restrict__ wvec,
    float* __restrict__ out, int total4)
{
  const int i = blockIdx.x * blockDim.x + threadIdx.x;
  if (i >= total4) return;
  const int base = i * 4;
  const int d = base & (DIM - 1);
  const int b = base / (NQ * DIM);
  const float4 qv = reinterpret_cast<const float4*>(q)[i];
  const float4 wv = *reinterpret_cast<const float4*>(wvec + b * DIM + d);
  float4 o;
  o.x = qv.x + wv.x; o.y = qv.y + wv.y;
  o.z = qv.z + wv.z; o.w = qv.w + wv.w;
  reinterpret_cast<float4*>(out)[i] = o;
}

extern "C" void kernel_launch(void* const* d_in, const int* in_sizes, int n_in,
                              void* d_out, int out_size, void* d_ws, size_t ws_size,
                              hipStream_t stream) {
  const float* queries = (const float*)d_in[0];   // (B,NQ,256)
  const float* navi    = (const float*)d_in[1];   // (B,2)
  const float* bev     = (const float*)d_in[2];   // (B,64,H,W)
  const int*   shp     = (const int*)d_in[3];     // [H,W]
  const float* pscore  = (const float*)d_in[4];   // (B,2)
  // d_in[5]=aw_w, d_in[6]=aw_b: unused -> softmax over a size-1 axis == 1.0
  const float* aws_w   = (const float*)d_in[7];   // (1,256)
  const float* aws_b   = (const float*)d_in[8];   // (1,)
  const float* conv_w  = (const float*)d_in[9];   // (256,64,3,3)
  const float* conv_b  = (const float*)d_in[10];  // (256,)
  const float* out_w   = (const float*)d_in[11];  // (256,256)
  const float* out_b   = (const float*)d_in[12];  // (256,)
  float* out = (float*)d_out;

  const int B = in_sizes[1] / 2;                  // navi_points is (B,2)

  float* vin  = (float*)d_ws;                     // B*256 floats
  float* wvec = (float*)d_ws + (size_t)B * DIM;   // B*256 floats

  // Stage 1: per-batch sample vector (tiny)
  gscba_stage1_sample<<<B, 256, 0, stream>>>(
      navi, bev, shp, pscore, aws_w, aws_b, conv_w, conv_b, vin);

  // Stage 2: 256x256xB matvec via WMMA (tiny)
  gscba_stage2_matvec_wmma<<<DIM / 16, 32, 0, stream>>>(out_w, out_b, vin,
                                                        wvec, B);

  // Stage 3: streaming add (the only HBM-bound part)
  const int total4 = (B * NQ * DIM) / 4;
  gscba_stage3_add<<<(total4 + 255) / 256, 256, 0, stream>>>(queries, wvec,
                                                             out, total4);
}
